// ComponentBasedHierarchicalAttention_58540404244694
// MI455X (gfx1250) — matrix operand order repaired and checked by reading, not compiled, not run on hardware
//
#include <hip/hip_runtime.h>
#include <hip/hip_bf16.h>

typedef __bf16 bf16_t;
typedef __attribute__((ext_vector_type(16))) __bf16 v16bf;
typedef __attribute__((ext_vector_type(8)))  float  v8f;
typedef __attribute__((ext_vector_type(4)))  __bf16 v4bf;
typedef __attribute__((ext_vector_type(4)))  int    v4i;

typedef __attribute__((address_space(1))) v4i* gv4i_ptr;   // global 128-bit chunk
typedef __attribute__((address_space(3))) v4i* lv4i_ptr;   // LDS 128-bit destination

#if defined(__has_builtin)
#if __has_builtin(__builtin_amdgcn_global_load_async_to_lds_b128) && \
    __has_builtin(__builtin_amdgcn_s_wait_asynccnt)
#define USE_ASYNC_LDS 1
#endif
#endif
#ifndef USE_ASYNC_LDS
#define USE_ASYNC_LDS 0
#endif

// ---------------------------------------------------------------------------
// f32 -> bf16 conversion (vectorized: 4 floats -> 4 bf16 per thread)
// ---------------------------------------------------------------------------
__global__ __launch_bounds__(256) void cvt_f32_bf16(const float4* __restrict__ in,
                                                    v4bf* __restrict__ out, int n4) {
    int i = blockIdx.x * blockDim.x + threadIdx.x;
    if (i < n4) {
        float4 f = in[i];
        v4bf o;
        o[0] = (__bf16)f.x; o[1] = (__bf16)f.y; o[2] = (__bf16)f.z; o[3] = (__bf16)f.w;
        out[i] = o;
    }
}

// ---------------------------------------------------------------------------
// Generic NT GEMM: Out[M,N] = A[M,K] * W[N,K]^T (+ bias[n])
// A and W row-major, K contiguous (lda = ldw = K).
// Block tile 64 x TN (TN = 64 or 128), 4 waves (128 threads),
// wave tile 32 x (TN/2): 2 A-frags x (TN/32) B-frags WMMA products.
// K-step 32, double-buffered LDS, rows padded to 40 bf16 (bank-conflict-free
// per-lane fragment gathers).  Global->LDS staging uses the CDNA5 async path
// (GLOBAL_LOAD_ASYNC_TO_LDS_B128 + s_wait_asynccnt) when available.
//
// OMODE: 0 = bf16 normal [M,N]   1 = bf16 transposed [N,M]
//        2 = f32  normal [M,N]   3 = f32  transposed [N,M]
// z = blockIdx.z; b = z / ncomp; i = z % ncomp; operand offsets b*s?b + i*s?i.
// ---------------------------------------------------------------------------
union FragU { v16bf v; uint4 u[2]; };

template <int OMODE, int TN>
__global__ void __launch_bounds__(128) gemm_nt(
    const bf16_t* __restrict__ A, long sAb, long sAi,
    const bf16_t* __restrict__ W, long sWb, long sWi,
    const float*  __restrict__ bias, long sBi,
    void* __restrict__ OutP, long sOb, long sOi, int ldo,
    int K, int ncomp)
{
    constexpr int GB = TN / 32;            // B frags per wave; B rows per thread
    __shared__ bf16_t ldsA[2][64 * 40];
    __shared__ bf16_t ldsB[2][TN * 40];

    const int z  = blockIdx.z;
    const int bb = z / ncomp;
    const int ci = z - bb * ncomp;
    const bf16_t* Ab = A + (long)bb * sAb + (long)ci * sAi;
    const bf16_t* Wb = W + (long)bb * sWb + (long)ci * sWi;
    const float* biasP = bias ? (bias + (long)ci * sBi) : nullptr;

    const int Mbase = blockIdx.x * 64;
    const int Nbase = blockIdx.y * TN;

    const int t  = threadIdx.x;
    const int r0 = t >> 2;        // staging row 0..31 (+32*rr)
    const int ch = (t & 3) * 8;   // staging K-chunk: 0,8,16,24

    const int lane = t & 31;
    const int wave = t >> 5;
    const int wm = wave >> 1;     // 0..1 : wave row in block tile
    const int wn = wave & 1;      // 0..1 : wave col in block tile
    const int l16 = lane & 15;
    const int hh  = lane >> 4;    // lane half

    v8f acc[2][GB] = {};
    const int nK = K >> 5;

#if USE_ASYNC_LDS
    auto stage = [&](int kt, int buf) {
        const long k0 = ((long)kt << 5) + ch;
#pragma unroll
        for (int rr = 0; rr < 2; ++rr) {
            const bf16_t* gp = Ab + (long)(Mbase + r0 + rr * 32) * K + k0;
            __builtin_amdgcn_global_load_async_to_lds_b128(
                (gv4i_ptr)(void*)gp,
                (lv4i_ptr)(void*)&ldsA[buf][(r0 + rr * 32) * 40 + ch], 0, 0);
        }
#pragma unroll
        for (int rr = 0; rr < GB; ++rr) {
            const bf16_t* gp = Wb + (long)(Nbase + r0 + rr * 32) * K + k0;
            __builtin_amdgcn_global_load_async_to_lds_b128(
                (gv4i_ptr)(void*)gp,
                (lv4i_ptr)(void*)&ldsB[buf][(r0 + rr * 32) * 40 + ch], 0, 0);
        }
    };
    stage(0, 0);
    __builtin_amdgcn_s_wait_asynccnt(0);
    __syncthreads();
#else
    uint4 rav[2], rbv[GB];
    auto gload = [&](int kt) {
        const long k0 = ((long)kt << 5) + ch;
#pragma unroll
        for (int rr = 0; rr < 2; ++rr)
            rav[rr] = *(const uint4*)(Ab + (long)(Mbase + r0 + rr * 32) * K + k0);
#pragma unroll
        for (int rr = 0; rr < GB; ++rr)
            rbv[rr] = *(const uint4*)(Wb + (long)(Nbase + r0 + rr * 32) * K + k0);
    };
    auto lstore = [&](int buf) {
#pragma unroll
        for (int rr = 0; rr < 2; ++rr)
            *(uint4*)&ldsA[buf][(r0 + rr * 32) * 40 + ch] = rav[rr];
#pragma unroll
        for (int rr = 0; rr < GB; ++rr)
            *(uint4*)&ldsB[buf][(r0 + rr * 32) * 40 + ch] = rbv[rr];
    };
    gload(0);
    lstore(0);
    __syncthreads();
#endif

    for (int kt = 0; kt < nK; ++kt) {
        const int  cur  = kt & 1;
        const bool more = (kt + 1) < nK;
#if USE_ASYNC_LDS
        if (more) stage(kt + 1, cur ^ 1);   // async DMA into the other buffer
#else
        if (more) gload(kt + 1);            // prefetch into registers
#endif

        // A fragment per ISA 7.12.2 (16-bit 16x32): lane half h -> K pairs at
        // 8h..8h+7 and 16+8h..23+8h.  B fragment (32x16): lane half h ->
        // same K order as A: 8h..8h+7 and 16+8h..23+8h.
        FragU fa[2], fb[GB];
#pragma unroll
        for (int f = 0; f < 2; ++f) {
            const int row = wm * 32 + f * 16 + l16;
            fa[f].u[0] = *(const uint4*)&ldsA[cur][row * 40 + 8 * hh];
            fa[f].u[1] = *(const uint4*)&ldsA[cur][row * 40 + 16 + 8 * hh];
        }
#pragma unroll
        for (int g = 0; g < GB; ++g) {
            const int row = wn * (TN / 2) + g * 16 + l16;
            fb[g].u[0] = *(const uint4*)&ldsB[cur][row * 40 + 8 * hh];
            fb[g].u[1] = *(const uint4*)&ldsB[cur][row * 40 + 16 + 8 * hh];
        }
#pragma unroll
        for (int f = 0; f < 2; ++f)
#pragma unroll
            for (int g = 0; g < GB; ++g)
                acc[f][g] = __builtin_amdgcn_wmma_f32_16x16x32_bf16(
                    false, fa[f].v, false, fb[g].v, (short)0, acc[f][g], false, false);

#if USE_ASYNC_LDS
        if (more) __builtin_amdgcn_s_wait_asynccnt(0);
#else
        if (more) lstore(cur ^ 1);
#endif
        __syncthreads();
    }

    // Epilogue.  C/D frag layout: n = lane%16, m = r + 8*(lane/16).
    const long zOff = (long)bb * sOb + (long)ci * sOi;
#pragma unroll
    for (int f = 0; f < 2; ++f) {
#pragma unroll
        for (int g = 0; g < GB; ++g) {
            const int col    = Nbase + wn * (TN / 2) + g * 16 + l16;
            const int mstart = Mbase + wm * 32 + f * 16 + 8 * hh;
            const float bv   = biasP ? biasP[col] : 0.f;
            if constexpr (OMODE == 0) {
                bf16_t* O = (bf16_t*)OutP + zOff;
#pragma unroll
                for (int r = 0; r < 8; ++r)
                    O[(long)(mstart + r) * ldo + col] = (bf16_t)(acc[f][g][r] + bv);
            } else if constexpr (OMODE == 1) {
                bf16_t* O = (bf16_t*)OutP + zOff;
                union { uint4 u; bf16_t e[8]; } pk;
#pragma unroll
                for (int r = 0; r < 8; ++r) pk.e[r] = (bf16_t)(acc[f][g][r] + bv);
                *(uint4*)&O[(long)col * ldo + mstart] = pk.u;   // 16B store
            } else if constexpr (OMODE == 2) {
                float* O = (float*)OutP + zOff;
#pragma unroll
                for (int r = 0; r < 8; ++r)
                    O[(long)(mstart + r) * ldo + col] = acc[f][g][r] + bv;
            } else {
                float* O = (float*)OutP + zOff;
                float4 u0 = make_float4(acc[f][g][0] + bv, acc[f][g][1] + bv,
                                        acc[f][g][2] + bv, acc[f][g][3] + bv);
                float4 u1 = make_float4(acc[f][g][4] + bv, acc[f][g][5] + bv,
                                        acc[f][g][6] + bv, acc[f][g][7] + bv);
                *(float4*)&O[(long)col * ldo + mstart]     = u0;
                *(float4*)&O[(long)col * ldo + mstart + 4] = u1;
            }
        }
    }
}

// ---------------------------------------------------------------------------
// Row softmax over 576 elements, one wave per row (18 elems/lane), bf16 out.
// ---------------------------------------------------------------------------
__global__ __launch_bounds__(256) void softmax576(const float* __restrict__ E,
                                                  bf16_t* __restrict__ A) {
    const int row  = blockIdx.x * 8 + (threadIdx.x >> 5);
    const int lane = threadIdx.x & 31;
    const float* e = E + (long)row * 576;
    bf16_t*      a = A + (long)row * 576;
    float v[18];
    float mx = -3.4e38f;
#pragma unroll
    for (int j = 0; j < 18; ++j) { v[j] = e[lane + 32 * j]; mx = fmaxf(mx, v[j]); }
#pragma unroll
    for (int o = 16; o > 0; o >>= 1) mx = fmaxf(mx, __shfl_xor(mx, o, 32));
    float s = 0.f;
#pragma unroll
    for (int j = 0; j < 18; ++j) { v[j] = __expf(v[j] - mx); s += v[j]; }
#pragma unroll
    for (int o = 16; o > 0; o >>= 1) s += __shfl_xor(s, o, 32);
    const float inv = 1.f / s;
#pragma unroll
    for (int j = 0; j < 18; ++j) a[lane + 32 * j] = (bf16_t)(v[j] * inv);
}

// ---------------------------------------------------------------------------
// combined[b][n][d] = x[b][n][d] + gamma * sum_i attn_outs[i][b][d][n]
// 64x64 LDS tile transpose so both sides stay coalesced.
// ---------------------------------------------------------------------------
__global__ __launch_bounds__(256) void combine_kernel(const float* __restrict__ AO,
                                                      const float* __restrict__ x,
                                                      const float* __restrict__ gamma,
                                                      float* __restrict__ out) {
    __shared__ float tbuf[64][65];
    const int b  = blockIdx.z;
    const int nt = blockIdx.y * 64;
    const int dt = blockIdx.x * 64;
    const long S = 4608L * 576;
    const int tid = threadIdx.x;
    const float g = gamma[0];

    const int nn = tid & 63;
    const int d0 = tid >> 6;
#pragma unroll
    for (int it = 0; it < 16; ++it) {
        const int dd = d0 + it * 4;
        const long base = ((long)b * 4608 + dt + dd) * 576 + nt + nn;
        tbuf[nn][dd] = AO[base] + AO[base + 8 * S] + AO[base + 16 * S];
    }
    __syncthreads();
    const int dd2 = tid & 63;
    const int n0  = tid >> 6;
#pragma unroll
    for (int it = 0; it < 16; ++it) {
        const int nn2 = n0 + it * 4;
        const long o = ((long)b * 576 + nt + nn2) * 4608 + dt + dd2;
        out[o] = x[o] + g * tbuf[nn2][dd2];
    }
}

// ---------------------------------------------------------------------------
extern "C" void kernel_launch(void* const* d_in, const int* in_sizes, int n_in,
                              void* d_out, int out_size, void* d_ws, size_t ws_size,
                              hipStream_t stream) {
    (void)in_sizes; (void)n_in; (void)out_size; (void)ws_size;
    const float* x     = (const float*)d_in[0];
    const float* Wq    = (const float*)d_in[1];
    const float* bq    = (const float*)d_in[2];
    const float* Wk    = (const float*)d_in[3];
    const float* bk    = (const float*)d_in[4];
    const float* Wv    = (const float*)d_in[5];
    const float* bv    = (const float*)d_in[6];
    const float* Wc    = (const float*)d_in[7];
    const float* bc    = (const float*)d_in[8];
    const float* gamma = (const float*)d_in[9];
    float* out = (float*)d_out;

    const long NX  = 8L * 576 * 4608;   // x / v-sized tensors
    const long NWQ = 576L * 4608;       // Wq / Wk
    const long NWV = 4608L * 4608;      // Wv
    const long NWC = 3L * 576 * 576;    // Wc
    const long NQ  = 8L * 576 * 576;    // q / kT

    bf16_t* xb  = (bf16_t*)d_ws;
    bf16_t* wqb = xb  + NX;
    bf16_t* wkb = wqb + NWQ;
    bf16_t* wvb = wkb + NWQ;
    bf16_t* wcb = wvb + NWV;
    bf16_t* qb  = wcb + NWC;
    bf16_t* ktb = qb  + NQ;
    bf16_t* vtb = ktb + NQ;
    bf16_t* cqb = vtb + NX;             // later reused as attn (bf16)
    float*  energy = (float*)wqb;       // aliases Wq/Wk/Wv (dead when written)

    auto cvt = [&](const float* src, bf16_t* dst, long n) {
        const int n4 = (int)(n >> 2);
        cvt_f32_bf16<<<dim3((n4 + 255) / 256), dim3(256), 0, stream>>>(
            (const float4*)src, (v4bf*)dst, n4);
    };
    cvt(x, xb, NX);  cvt(Wq, wqb, NWQ); cvt(Wk, wkb, NWQ);
    cvt(Wv, wvb, NWV); cvt(Wc, wcb, NWC);

    const dim3 blk(128);
    // q = xt @ Wq^T + bq           -> qb [b][n][k]      (bf16, normal)
    gemm_nt<0, 64><<<dim3(9, 9, 8), blk, 0, stream>>>(
        xb, 576L * 4608, 0, wqb, 0, 0, bq, 0,
        qb, 576L * 576, 0, 576, 4608, 1);
    // kT = (xt @ Wk^T + bk)^T      -> ktb [b][c8][n]    (bf16, transposed)
    gemm_nt<1, 64><<<dim3(9, 9, 8), blk, 0, stream>>>(
        xb, 576L * 4608, 0, wkb, 0, 0, bk, 0,
        ktb, 576L * 576, 0, 576, 4608, 1);
    // vT = (xt @ Wv^T + bv)^T      -> vtb [b][d][n]     (bf16, transposed, TN=128)
    gemm_nt<1, 128><<<dim3(9, 36, 8), blk, 0, stream>>>(
        xb, 576L * 4608, 0, wvb, 0, 0, bv, 0,
        vtb, 4608L * 576, 0, 576, 4608, 1);
    // cq = q @ Wc[i]^T + bc[i]     -> cqb [b*3+i][n][m] (bf16, normal)
    gemm_nt<0, 64><<<dim3(9, 9, 24), blk, 0, stream>>>(
        qb, 576L * 576, 0, wcb, 0, 576L * 576, bc, 576,
        cqb, 3L * 576 * 576, 576L * 576, 576, 576, 3);
    // energy = cq @ k              -> energy [b*3+i][n][c] (f32, normal)
    gemm_nt<2, 64><<<dim3(9, 9, 24), blk, 0, stream>>>(
        cqb, 3L * 576 * 576, 576L * 576, ktb, 576L * 576, 0, nullptr, 0,
        energy, 3L * 576 * 576, 576L * 576, 576, 576, 3);
    // attn = softmax(energy)       -> cqb (bf16, reuse)
    softmax576<<<dim3(24 * 576 / 8), dim3(256), 0, stream>>>(energy, cqb);
    // out_i = attn @ v  -> d_out attn_outs region [i][b][d][n] (f32, transposed, TN=128)
    float* attnOut = out + NX;
    gemm_nt<3, 128><<<dim3(9, 36, 24), blk, 0, stream>>>(
        cqb, 3L * 576 * 576, 576L * 576, vtb, 4608L * 576, 0, nullptr, 0,
        attnOut, 4608L * 576, 8L * 4608 * 576, 576, 576, 3);
    // combined = x + gamma * sum_i out_i
    combine_kernel<<<dim3(72, 9, 8), dim3(256), 0, stream>>>(attnOut, x, gamma, out);
}
